// SpikingLayer_41489384080071
// MI455X (gfx1250) — compile-verified, matrix operand order checked by reading, not run on hardware
//
#include <hip/hip_runtime.h>
#include <hip/hip_bf16.h>
#include <math.h>

// ---------------------------------------------------------------------------
// SpikingLayer (SLSTM) for MI455X / gfx1250, wave32 + WMMA f16 + TDM.
//   B=32, S=2048, D=1024, gates=4D=4096
//
// Recurrence is the bottleneck (serial over S=2048 steps; HBM floor is only
// ~35us, so we optimize the per-step critical path):
//   - 16 persistent WGs = 2 batch-groups (16 rows) x 8 column-groups (128 D).
//     Batch rows are independent -> each group has its OWN barrier counter,
//     halving the coupled WG set and the per-wave serial WMMA count (256/step).
//   - Gates via v_wmma_f32_16x16x32_f16; W_ih/W_hh stay f16 in L2 (16 MB).
//   - x_t / mem staged to LDS per step by the Tensor Data Mover
//     (tensor_load_to_lds + s_wait_tensorcnt) when the builtin exists;
//     synchronous v8h copies otherwise.
//   - LSTM elementwise runs on accumulator fragments in registers; syn/mem
//     state lives in registers for all 2048 steps; mem double-buffered in L2.
// Then a tiled WMMA GEMM fuses fc + bias + residual + DyT epilogue.
// ---------------------------------------------------------------------------

typedef _Float16 v8h  __attribute__((ext_vector_type(8)));
typedef _Float16 v16h __attribute__((ext_vector_type(16)));
typedef float    v8f  __attribute__((ext_vector_type(8)));
typedef unsigned int v4u __attribute__((ext_vector_type(4)));
typedef int      v4i  __attribute__((ext_vector_type(4)));
typedef int      v8i  __attribute__((ext_vector_type(8)));

#define B_    32
#define S_    2048
#define D_    1024
#define G4_   4096
#define GRPW_ 8      // WGs per batch group
#define NWG_  16     // total persistent WGs (2 groups x 8)

#if defined(__has_builtin)
#  if __has_builtin(__builtin_amdgcn_tensor_load_to_lds) && \
      __has_builtin(__builtin_amdgcn_s_wait_tensorcnt)
#    define HAVE_TDM 1
#  endif
#endif
#ifndef HAVE_TDM
#  define HAVE_TDM 0
#endif

__device__ __forceinline__ v8f wmma16(v16h a, v16h b, v8f c) {
  return __builtin_amdgcn_wmma_f32_16x16x32_f16(
      /*neg_a=*/false, a, /*neg_b=*/false, b,
      /*c_mod=*/(short)0, c, /*reuse_a=*/false, /*reuse_b=*/false);
}

// 16x32 (A, lane=M) / 32x16 (B, lane=N) f16 fragment from row-major memory.
// ISA 7.12.2: lanes 0-15 row=row0+lane, halves K k0..k0+7 / k0+16..k0+23;
//             lanes 16-31 row=row0+lane-16, halves K k0+8.. / k0+24..
__device__ __forceinline__ v16h load_frag16(const _Float16* base, int row0,
                                            int stride, int k0, int lane) {
  const int r  = row0 + (lane & 15);
  const int kk = k0 + ((lane >> 4) << 3);
  const _Float16* p = base + (size_t)r * (size_t)stride + kk;
  v8h lo = *(const v8h*)(p);
  v8h hi = *(const v8h*)(p + 16);
  v16h o;
#pragma unroll
  for (int i = 0; i < 8; ++i) { o[i] = lo[i]; o[i + 8] = hi[i]; }
  return o;
}

__device__ __forceinline__ float sigmoidf_(float x) {
  return 1.0f / (1.0f + __expf(-x));
}

#if HAVE_TDM
// 1D contiguous copy of `nelem` 2-byte elements, global -> LDS byte offset.
// D# per ISA 08_async_tensor: group0 = {count=1 | lds_addr | global_addr[56:0]
// | type=2}; group1 = {data_size=2B, tensor_dim0=tile_dim0=nelem, tile_dim1=1}.
__device__ __forceinline__ void tdm_load_2b(unsigned lds_byte_off,
                                            const void* gsrc, int nelem) {
  unsigned long long ga = (unsigned long long)(uintptr_t)gsrc;
  v4u g0 = {1u,                                   // count=1 (valid user D#)
            lds_byte_off,                          // lds_addr
            (unsigned)(ga & 0xffffffffu),          // global_addr[31:0]
            (unsigned)((ga >> 32) & 0x01ffffffu) | (2u << 30)};  // [56:32]|type=2
  unsigned n = (unsigned)nelem;
  v8i g1 = {(int)(1u << 16),                      // wg_mask=0, data_size=1 (2B)
            (int)((n & 0xffffu) << 16),           // tensor_dim0[15:0]
            (int)(((n >> 16) & 0xffffu) | (1u << 16)),  // dim0[31:16], dim1=1
            (int)((n & 0xffffu) << 16),           // tile_dim0 = n (16-bit)
            1,                                    // tile_dim1 = 1
            (int)n,                               // tensor_dim0_stride[31:0]
            0, 0};
  v4i z4 = {0, 0, 0, 0};
#if __clang_major__ >= 23
  v8i z8 = {0, 0, 0, 0, 0, 0, 0, 0};
  __builtin_amdgcn_tensor_load_to_lds(g0, g1, z4, z4, z8, 0);
#else
  __builtin_amdgcn_tensor_load_to_lds(g0, g1, z4, z4, 0);
#endif
}
#endif

// ------------------------------ convert kernels ----------------------------

__global__ void k_zero_bar(unsigned* bar) {
  if (blockIdx.x == 0 && threadIdx.x < 128) bar[threadIdx.x] = 0u;
}

__global__ void k_cvt_f16(const float* __restrict__ src,
                          _Float16* __restrict__ dst, int n) {
  int i = blockIdx.x * blockDim.x + threadIdx.x;
  if (i < n) dst[i] = (_Float16)src[i];
}

// x [B][S][D] fp32 -> xh [S][B][D] f16 (x_t becomes a contiguous slab)
__global__ void k_cvt_x_tr(const float* __restrict__ x,
                           _Float16* __restrict__ xh) {
  size_t i  = (size_t)blockIdx.x * blockDim.x + threadIdx.x;  // < B*S*D
  size_t d  = i & (D_ - 1);
  size_t bs = i >> 10;            // b*S + s
  size_t s  = bs & (S_ - 1);
  size_t b  = bs >> 11;
  xh[((s * B_ + b) << 10) | d] = (_Float16)x[i];
}

// ------------------------------ recurrent kernel ---------------------------
// grid = 16 WGs: group = blockIdx.x & 1 (batch rows group*16..+15),
//                colwg = blockIdx.x >> 1 (D columns colwg*128..+127).
// wave owns 16 D-columns x all 4 gates x 1 m-tile -> elementwise in registers.

__global__ __launch_bounds__(256) void snn_recurrent(
    const _Float16* __restrict__ xh,    // [S][B][D] f16
    const _Float16* __restrict__ wih,   // [4D][D] f16 row-major
    const _Float16* __restrict__ whh,   // [4D][D] f16 row-major
    const float* __restrict__ b_ih, const float* __restrict__ b_hh,
    const float* __restrict__ thr_p,
    const float* __restrict__ syn0, const float* __restrict__ mem0,
    _Float16* __restrict__ memg,        // [2][B][D] f16 double buffer
    _Float16* __restrict__ spk,         // [B*S][D] f16, row = b*S + t
    float* __restrict__ syn_out,        // [B][D] fp32 (d_out tail)
    float* __restrict__ mem_out,        // [B][D] fp32 (d_out tail)
    unsigned* __restrict__ bar) {       // bar[group*64]
  extern __shared__ _Float16 smem_dyn[];      // 64 KB dynamic LDS
  _Float16* xs = smem_dyn;                    // [16][D] x_t   (group rows)
  _Float16* ms = smem_dyn + 16 * D_;          // [16][D] mem_p (group rows)

  const int tid   = threadIdx.x;
  const int lane  = tid & 31;
  const int wave  = tid >> 5;
  const int group = blockIdx.x & 1;            // batch-row group
  const int colwg = blockIdx.x >> 1;           // column group
  const int d0    = colwg * 128 + wave * 16;   // this wave's 16 D-columns
  const int col   = d0 + (lane & 15);
  const int hi8   = (lane >> 4) << 3;
  const int rowb0 = group * 16;                // first batch row of group
  const float thr = thr_p[0];
  unsigned* mybar = bar + group * 64;          // 256B-separated counters

  const float bias_i = b_ih[col]          + b_hh[col];
  const float bias_f = b_ih[D_ + col]     + b_hh[D_ + col];
  const float bias_g = b_ih[2 * D_ + col] + b_hh[2 * D_ + col];
  const float bias_o = b_ih[3 * D_ + col] + b_hh[3 * D_ + col];

  // persistent cell state: element r -> (batch rowb0 + r + hi8, col)
  v8f syn_fr, mem_fr;
#pragma unroll
  for (int r = 0; r < 8; ++r) {
    int rb = rowb0 + r + hi8;
    syn_fr[r] = syn0[(size_t)rb * D_ + col];
    mem_fr[r] = mem0[(size_t)rb * D_ + col];
  }

  for (int t = 0; t < S_; ++t) {
    const _Float16* xt = xh + ((size_t)t * B_ + rowb0) * D_;
    const _Float16* mg = memg + ((size_t)(t & 1) * B_ + rowb0) * D_;
#if HAVE_TDM
    if (wave == 0) {   // one TDM descriptor per 32 KB slab, EXEC-independent
      tdm_load_2b(0u, xt, 16 * D_);
      tdm_load_2b((unsigned)(16 * D_ * sizeof(_Float16)), mg, 16 * D_);
      __builtin_amdgcn_s_wait_tensorcnt(0);
    }
    __syncthreads();
#else
#pragma unroll 4
    for (int i = tid; i < (16 * D_) / 8; i += 256) {
      ((v8h*)xs)[i] = ((const v8h*)xt)[i];
      ((v8h*)ms)[i] = ((const v8h*)mg)[i];
    }
    __syncthreads();
#endif

    v8f acc[4];   // i, f, g, o
#pragma unroll
    for (int r = 0; r < 8; ++r) {
      acc[0][r] = bias_i; acc[1][r] = bias_f;
      acc[2][r] = bias_g; acc[3][r] = bias_o;
    }

    for (int k0 = 0; k0 < D_; k0 += 32) {
      v16h ax = load_frag16(xs, 0, D_, k0, lane);
      v16h am = load_frag16(ms, 0, D_, k0, lane);
#pragma unroll
      for (int g = 0; g < 4; ++g) {
        v16h bi = load_frag16(wih, g * D_ + d0, D_, k0, lane);  // L2-resident
        v16h bh = load_frag16(whh, g * D_ + d0, D_, k0, lane);
        acc[g] = wmma16(ax, bi, acc[g]);
        acc[g] = wmma16(am, bh, acc[g]);
      }
    }

    // LSTM elementwise + spike, in registers
    _Float16* mdst = memg + (size_t)((t + 1) & 1) * (B_ * D_);
#pragma unroll
    for (int r = 0; r < 8; ++r) {
      float ig = sigmoidf_(acc[0][r]);
      float fg = sigmoidf_(acc[1][r]);
      float gg = tanhf(acc[2][r]);
      float og = sigmoidf_(acc[3][r]);
      float reset = (mem_fr[r] > thr) ? 1.0f : 0.0f;  // heaviside(mem_p - thr)
      float c  = fg * syn_fr[r] + ig * gg;
      float h  = og * tanhf(c);
      float mn = h - reset * thr;
      float sp = (mn > thr) ? 1.0f : 0.0f;            // heaviside(mem_new-thr)
      syn_fr[r] = c;
      mem_fr[r] = mn;
      int rb = rowb0 + r + hi8;
      mdst[(size_t)rb * D_ + col] = (_Float16)mn;
      spk[((size_t)rb * S_ + t) * D_ + col] = (_Float16)sp;
    }

    // barrier across this batch group's 8 WGs only
    __threadfence();
    __syncthreads();
    if (tid == 0) {
      atomicAdd(mybar, 1u);
      const unsigned target = (unsigned)(t + 1) * GRPW_;
      while (__hip_atomic_load(mybar, __ATOMIC_RELAXED,
                               __HIP_MEMORY_SCOPE_AGENT) < target) { }
    }
    __syncthreads();
    __threadfence();
  }

#pragma unroll
  for (int r = 0; r < 8; ++r) {
    int rb = rowb0 + r + hi8;
    syn_out[(size_t)rb * D_ + col] = syn_fr[r];
    mem_out[(size_t)rb * D_ + col] = mem_fr[r];
  }
}

// ------------------------- fc GEMM + residual + DyT ------------------------
// out[r,:] = gamma * tanh(alpha * (spk[r,:]@fc_w^T + fc_b + x[r,:])) + beta

__global__ __launch_bounds__(256) void fc_dyt(
    const _Float16* __restrict__ spk,   // [B*S][D] f16
    const _Float16* __restrict__ fcw,   // [D][D] f16 row-major
    const float* __restrict__ fcb,
    const float* __restrict__ xin,      // [B*S][D] fp32 (original x, flat)
    const float* __restrict__ alpha_p,
    const float* __restrict__ gamma, const float* __restrict__ beta,
    float* __restrict__ out) {
  __shared__ _Float16 As[128 * 32];     // 8 KB
  __shared__ _Float16 Bs[64 * 32];      // 4 KB

  const int tid  = threadIdx.x;
  const int lane = tid & 31;
  const int wave = tid >> 5;
  const size_t row0 = (size_t)blockIdx.x * 128;
  const int n0 = blockIdx.y * 64;
  const int hi8 = (lane >> 4) << 3;

  v8f acc[4];
#pragma unroll
  for (int nt = 0; nt < 4; ++nt)
#pragma unroll
    for (int r = 0; r < 8; ++r) acc[nt][r] = 0.0f;

  for (int k0 = 0; k0 < D_; k0 += 32) {
    __syncthreads();
#pragma unroll
    for (int i = tid; i < (128 * 32) / 8; i += 256) {
      int rr = (i * 8) >> 5, cc = (i * 8) & 31;
      ((v8h*)As)[i] = *(const v8h*)(spk + (row0 + rr) * D_ + k0 + cc);
      if (k0 + 32 < D_)
        __builtin_prefetch(spk + (row0 + rr) * D_ + k0 + 32 + cc, 0, 1);
    }
#pragma unroll
    for (int i = tid; i < (64 * 32) / 8; i += 256) {
      int rr = (i * 8) >> 5, cc = (i * 8) & 31;
      ((v8h*)Bs)[i] = *(const v8h*)(fcw + (size_t)(n0 + rr) * D_ + k0 + cc);
    }
    __syncthreads();

    v16h a = load_frag16(As, wave * 16, 32, 0, lane);
#pragma unroll
    for (int nt = 0; nt < 4; ++nt) {
      v16h b = load_frag16(Bs, nt * 16, 32, 0, lane);
      acc[nt] = wmma16(a, b, acc[nt]);
    }
  }

  const float alpha = alpha_p[0];
#pragma unroll
  for (int nt = 0; nt < 4; ++nt) {
    const int colD = n0 + nt * 16 + (lane & 15);
    const float fb = fcb[colD], gm = gamma[colD], bt = beta[colD];
#pragma unroll
    for (int r = 0; r < 8; ++r) {
      size_t row = row0 + wave * 16 + r + hi8;
      float v  = acc[nt][r] + fb;
      float xv = xin[row * D_ + colD];
      out[row * D_ + colD] = gm * tanhf(alpha * (v + xv)) + bt;
    }
  }
}

// ------------------------------- launcher ----------------------------------

extern "C" void kernel_launch(void* const* d_in, const int* in_sizes, int n_in,
                              void* d_out, int out_size, void* d_ws,
                              size_t ws_size, hipStream_t stream) {
  const float* x      = (const float*)d_in[0];
  const float* syn0   = (const float*)d_in[1];
  const float* mem0   = (const float*)d_in[2];
  const float* w_ih   = (const float*)d_in[3];
  const float* w_hh   = (const float*)d_in[4];
  const float* b_ih   = (const float*)d_in[5];
  const float* b_hh   = (const float*)d_in[6];
  const float* thr    = (const float*)d_in[7];
  const float* fc_w   = (const float*)d_in[8];
  const float* fc_b   = (const float*)d_in[9];
  const float* dalpha = (const float*)d_in[10];
  const float* dgamma = (const float*)d_in[11];
  const float* dbeta  = (const float*)d_in[12];

  float* out     = (float*)d_out;
  float* syn_out = out + (size_t)B_ * S_ * D_;
  float* mem_out = syn_out + (size_t)B_ * D_;

  char* ws = (char*)d_ws;
  unsigned* bar  = (unsigned*)ws;                       // 512 B (2 counters)
  _Float16* xh   = (_Float16*)(ws + 512);               // 128 MB [S][B][D]
  _Float16* wihh = xh   + (size_t)S_ * B_ * D_;         // 8 MB
  _Float16* whhh = wihh + (size_t)G4_ * D_;             // 8 MB
  _Float16* fcwh = whhh + (size_t)G4_ * D_;             // 2 MB
  _Float16* memg = fcwh + (size_t)D_ * D_;              // 128 KB [2][B][D]
  _Float16* spk  = memg + (size_t)2 * B_ * D_;          // 128 MB [B*S][D]

  k_zero_bar<<<1, 128, 0, stream>>>(bar);
  k_cvt_x_tr<<<(B_ * S_ * D_) / 256, 256, 0, stream>>>(x, xh);
  k_cvt_f16<<<(G4_ * D_) / 256, 256, 0, stream>>>(w_ih, wihh, G4_ * D_);
  k_cvt_f16<<<(G4_ * D_) / 256, 256, 0, stream>>>(w_hh, whhh, G4_ * D_);
  k_cvt_f16<<<(D_ * D_) / 256, 256, 0, stream>>>(fc_w, fcwh, D_ * D_);
  k_cvt_f16<<<(B_ * D_) / 256, 256, 0, stream>>>(mem0, memg, B_ * D_);

  const int smem_bytes = 2 * 16 * D_ * (int)sizeof(_Float16);  // 64 KB
  (void)hipFuncSetAttribute((const void*)snn_recurrent,
                            hipFuncAttributeMaxDynamicSharedMemorySize,
                            smem_bytes);
  snn_recurrent<<<NWG_, 256, smem_bytes, stream>>>(
      xh, wihh, whhh, b_ih, b_hh, thr, syn0, mem0, memg, spk, syn_out, mem_out,
      bar);

  fc_dyt<<<dim3((B_ * S_) / 128, D_ / 64), 256, 0, stream>>>(
      spk, fcwh, fc_b, x, dalpha, dgamma, dbeta, out);
}
